// DirectMPIGOGate_11544872091650
// MI455X (gfx1250) — matrix operand order repaired and checked by reading, not compiled, not run on hardware
//
#include <hip/hip_runtime.h>

typedef __attribute__((ext_vector_type(16))) _Float16 v16h;
typedef __attribute__((ext_vector_type(8)))  float    v8f;

#define M_PTS   1048576
#define NCH     12
#define GRID_N  128
#define WID     64
#define EOUT    8
#define DIM0    39

static __device__ inline v8f wmma_f16(v16h a, v16h b, v8f c) {
  // D = A(16x32 f16) * B(32x16 f16) + C(16x16 f32)
  return __builtin_amdgcn_wmma_f32_16x16x32_f16(false, a, false, b, (short)0, c,
                                                false, false);
}

// A fragment, contiguous K order (same as B): lane holds row = lane&15;
// elements 0..7  -> K = kbase + 2*khalf + i      (khalf = (lane>>4)*8)
// elements 8..15 -> K = kbase + 2*khalf + 8 + i
static __device__ inline v16h load_afrag(const _Float16* __restrict__ tile,
                                         int row, int khalf, int kbase) {
  v16h a;
#pragma unroll
  for (int i = 0; i < 8; ++i) a[i]     = tile[row * 64 + kbase + 2 * khalf + i];
#pragma unroll
  for (int i = 0; i < 8; ++i) a[i + 8] = tile[row * 64 + kbase + 2 * khalf + 8 + i];
  return a;
}

__launch_bounds__(256)
__global__ void nerf_mlp_wmma(const float* __restrict__ ray_pts,
                              const float* __restrict__ viewdirs,
                              const float* __restrict__ k0,
                              const float* __restrict__ w0,
                              const float* __restrict__ b0,
                              const float* __restrict__ w1,
                              const float* __restrict__ b1,
                              const float* __restrict__ w2,
                              const float* __restrict__ b2,
                              const int*   __restrict__ ray_id,
                              float* __restrict__ out,
                              int nTiles) {
  __shared__ _Float16 w0L[64 * 64];       // 39x64 padded to 64x64
  __shared__ _Float16 w1L[64 * 64];       // 64x64
  __shared__ _Float16 w2L[64 * 16];       // 64x8 padded to 64x16
  __shared__ _Float16 featS[8][16 * 64];  // per-wave feature staging (pad written once)
  __shared__ _Float16 hS[8][16 * 64];     // per-wave hidden-activation staging

  const int tid   = threadIdx.x;
  const int lane  = tid & 31;
  const int wave  = tid >> 5;
  const int lo16  = lane & 15;
  const int hi    = lane >> 4;     // 0 or 1
  const int khalf = hi * 8;        // A-layout K offset
  const int khf2  = hi * 16;       // B-layout K offset

  // ---------------- weights -> LDS (f16, padded) ----------------
  for (int idx = tid; idx < 64 * 64; idx += 256) {
    int r = idx >> 6, c = idx & 63;
    w0L[idx] = (_Float16)(r < DIM0 ? w0[r * 64 + c] : 0.0f);
    w1L[idx] = (_Float16)w1[idx];
  }
  for (int idx = tid; idx < 64 * 16; idx += 256) {
    int r = idx >> 4, c = idx & 15;
    w2L[idx] = (_Float16)(c < EOUT ? w2[r * EOUT + c] : 0.0f);
  }
  __syncthreads();  // only cross-wave dependency in the kernel

  // ---------------- B fragments -> registers ----------------
  // B fragment, contiguous K order (same as A): lane holds col = lane&15; element i -> K = khf2 + i
  v16h B0[2][4], B1[2][4], B2[2];
#pragma unroll
  for (int ks = 0; ks < 2; ++ks) {
#pragma unroll
    for (int nt = 0; nt < 4; ++nt) {
      v16h b, b1f;
#pragma unroll
      for (int i = 0; i < 16; ++i) {
        b[i]   = w0L[(ks * 32 + khf2 + i) * 64 + nt * 16 + lo16];
        b1f[i] = w1L[(ks * 32 + khf2 + i) * 64 + nt * 16 + lo16];
      }
      B0[ks][nt] = b;
      B1[ks][nt] = b1f;
    }
    v16h b2f;
#pragma unroll
    for (int i = 0; i < 16; ++i)
      b2f[i] = w2L[(ks * 32 + khf2 + i) * 16 + lo16];
    B2[ks] = b2f;
  }

  float b0v[4], b1v[4];
#pragma unroll
  for (int nt = 0; nt < 4; ++nt) {
    b0v[nt] = b0[nt * 16 + lo16];
    b1v[nt] = b1[nt * 16 + lo16];
  }
  const float b2v = (lo16 < EOUT) ? b2[lo16] : 0.0f;

  _Float16* featL = featS[wave];
  _Float16* hL    = hS[wave];

  // one-time zero fill of the feature K padding (K = 39..63), wave-private
  {
    const int row = lane;  // 32 lanes cover rows 0..15 twice? no: rows 0..31 -> clamp
    if (lane < 16) {
#pragma unroll
      for (int k = DIM0; k < 64; ++k) featL[lane * 64 + k] = (_Float16)0.0f;
    }
    (void)row;
  }

  // 65536 tiles, 8 waves/block, grid=1024 -> 8 iterations per wave; waves
  // are fully independent (all staging is wave-private, lockstep wave32).
  for (int t = blockIdx.x * 8 + wave; t < nTiles; t += gridDim.x * 8) {
    const int base = t * 16;
    const int p    = base + lo16;  // this lane-pair's point

    // ------- stage features: feat[16][64] f16 (balanced lane pair) -------
    // Both halves compute the trilinear corner weights for their point.
    const float px = ray_pts[p * 3 + 0] * 127.0f;
    const float py = ray_pts[p * 3 + 1] * 127.0f;
    const float pz = ray_pts[p * 3 + 2] * 127.0f;
    const float flx = floorf(px), fly = floorf(py), flz = floorf(pz);
    const int ix = min(max((int)flx, 0), 127);
    const int iy = min(max((int)fly, 0), 127);
    const int iz = min(max((int)flz, 0), 127);
    const int jx = min(ix + 1, 127), jy = min(iy + 1, 127), jz = min(iz + 1, 127);
    const float fx = px - flx, fy = py - fly, fz = pz - flz;
    const float gx = 1.0f - fx, gy = 1.0f - fy, gz = 1.0f - fz;
    const int o000 = (ix * GRID_N + iy) * GRID_N + iz;
    const int o001 = (ix * GRID_N + iy) * GRID_N + jz;
    const int o010 = (ix * GRID_N + jy) * GRID_N + iz;
    const int o011 = (ix * GRID_N + jy) * GRID_N + jz;
    const int o100 = (jx * GRID_N + iy) * GRID_N + iz;
    const int o101 = (jx * GRID_N + iy) * GRID_N + jz;
    const int o110 = (jx * GRID_N + jy) * GRID_N + iz;
    const int o111 = (jx * GRID_N + jy) * GRID_N + jz;
    const float w000 = gx * gy * gz, w001 = gx * gy * fz;
    const float w010 = gx * fy * gz, w011 = gx * fy * fz;
    const float w100 = fx * gy * gz, w101 = fx * gy * fz;
    const float w110 = fx * fy * gz, w111 = fx * fy * fz;

    const int cBeg = hi ? 8 : 0;
    const int cEnd = hi ? NCH : 8;
#pragma unroll
    for (int c = 0; c < NCH; ++c) {
      if (c >= cBeg && c < cEnd) {
        const float* __restrict__ g = k0 + c * (GRID_N * GRID_N * GRID_N);
        float v = g[o000] * w000 + g[o001] * w001 + g[o010] * w010 +
                  g[o011] * w011 + g[o100] * w100 + g[o101] * w101 +
                  g[o110] * w110 + g[o111] * w111;
        featL[lo16 * 64 + c] = (_Float16)v;
      }
    }
    if (hi) {
      // view-direction positional encoding (per-ray, gathered by ray_id)
      const int rid = ray_id[p];
      float vdv[3];
      vdv[0] = viewdirs[rid * 3 + 0];
      vdv[1] = viewdirs[rid * 3 + 1];
      vdv[2] = viewdirs[rid * 3 + 2];
      featL[lo16 * 64 + 12] = (_Float16)vdv[0];
      featL[lo16 * 64 + 13] = (_Float16)vdv[1];
      featL[lo16 * 64 + 14] = (_Float16)vdv[2];
#pragma unroll
      for (int d = 0; d < 3; ++d) {
#pragma unroll
        for (int e = 0; e < 4; ++e) {
          const float ang = vdv[d] * (float)(1 << e);
          featL[lo16 * 64 + 15 + d * 4 + e] = (_Float16)__sinf(ang);
          featL[lo16 * 64 + 27 + d * 4 + e] = (_Float16)__cosf(ang);
        }
      }
    }
    // wave-private LDS: hardware keeps per-wave DS ops ordered; compiler
    // inserts s_wait_dscnt for the dependent loads below. No block barrier.

    // ---------------- layer 0: (16x64) @ (64x64) ----------------
    v16h a0 = load_afrag(featL, lo16, khalf, 0);
    v16h a1 = load_afrag(featL, lo16, khalf, 32);
    v8f acc[4];
#pragma unroll
    for (int nt = 0; nt < 4; ++nt) {
      v8f c = {};
      c = wmma_f16(a0, B0[0][nt], c);
      c = wmma_f16(a1, B0[1][nt], c);
      acc[nt] = c;
    }
    // bias + relu, restage as f16 A-input (D layout: row = r + hi*8, col = nt*16+lo16)
#pragma unroll
    for (int nt = 0; nt < 4; ++nt)
#pragma unroll
      for (int r = 0; r < 8; ++r) {
        float v = fmaxf(acc[nt][r] + b0v[nt], 0.0f);
        hL[(r + hi * 8) * 64 + nt * 16 + lo16] = (_Float16)v;
      }

    // ---------------- layer 1: (16x64) @ (64x64) ----------------
    a0 = load_afrag(hL, lo16, khalf, 0);
    a1 = load_afrag(hL, lo16, khalf, 32);
#pragma unroll
    for (int nt = 0; nt < 4; ++nt) {
      v8f c = {};
      c = wmma_f16(a0, B1[0][nt], c);
      c = wmma_f16(a1, B1[1][nt], c);
      acc[nt] = c;
    }
#pragma unroll
    for (int nt = 0; nt < 4; ++nt)
#pragma unroll
      for (int r = 0; r < 8; ++r) {
        float v = fmaxf(acc[nt][r] + b1v[nt], 0.0f);
        hL[(r + hi * 8) * 64 + nt * 16 + lo16] = (_Float16)v;
      }

    // ---------------- layer 2: (16x64) @ (64x16pad) ----------------
    a0 = load_afrag(hL, lo16, khalf, 0);
    a1 = load_afrag(hL, lo16, khalf, 32);
    v8f c2 = {};
    c2 = wmma_f16(a0, B2[0], c2);
    c2 = wmma_f16(a1, B2[1], c2);

    if (lo16 < EOUT) {
#pragma unroll
      for (int r = 0; r < 8; ++r)
        out[(size_t)(base + r + hi * 8) * EOUT + lo16] = c2[r] + b2v;
    }
  }
}

extern "C" void kernel_launch(void* const* d_in, const int* in_sizes, int n_in,
                              void* d_out, int out_size, void* d_ws, size_t ws_size,
                              hipStream_t stream) {
  const float* ray_pts  = (const float*)d_in[0];
  const float* viewdirs = (const float*)d_in[1];
  const float* k0       = (const float*)d_in[2];
  const float* w0       = (const float*)d_in[3];
  const float* b0       = (const float*)d_in[4];
  const float* w1       = (const float*)d_in[5];
  const float* b1       = (const float*)d_in[6];
  const float* w2       = (const float*)d_in[7];
  const float* b2       = (const float*)d_in[8];
  const int*   ray_id   = (const int*)d_in[9];
  float*       out      = (float*)d_out;

  const int nTiles = M_PTS / 16;  // 65536
  hipLaunchKernelGGL(nerf_mlp_wmma, dim3(1024), dim3(256), 0, stream,
                     ray_pts, viewdirs, k0, w0, b0, w1, b1, w2, b2, ray_id,
                     out, nTiles);
}
